// DepthConv_60687887892852
// MI455X (gfx1250) — matrix operand order repaired and checked by reading, not compiled, not run on hardware
//
#include <hip/hip_runtime.h>

typedef __attribute__((ext_vector_type(16))) __bf16        v16bf;
typedef __attribute__((ext_vector_type(8)))  float         v8f;
typedef __attribute__((ext_vector_type(8)))  unsigned int  v8u;
typedef __attribute__((ext_vector_type(4)))  unsigned int  v4u;

union frag16 {            // 32-byte 16-element bf16 fragment
  v16bf bf;
  v4u   q[2];
};

#define B_    8
#define CIN   64
#define COUT  64
#define H_    128
#define W_    128

// LDS layout (bytes)
#define WLDS_BYTES (9 * 64 * 64 * 2)        // bf16 weights [tap][o][c]   = 73728
#define XLDS_ELEMS (64 * 10 * 34)           // bf16 x tile  [hh][ww][c]   = 21760
#define XLDS_BYTES (XLDS_ELEMS * 2)         //                             = 43520
#define DLDS_ELEMS (10 * 34)                // f32 depth tile [hh][ww]
#define DLDS_BYTES (DLDS_ELEMS * 4)
#define BLDS_BYTES (64 * 4)
#define SMEM_BYTES (WLDS_BYTES + XLDS_BYTES + DLDS_BYTES + BLDS_BYTES)   // 118864

__global__ __launch_bounds__(256) void depthconv_wmma_bf16(
    const float* __restrict__ x, const float* __restrict__ depth,
    const float* __restrict__ weight, const float* __restrict__ bias,
    float* __restrict__ out) {
  extern __shared__ unsigned char smem[];
  __bf16* wlds = (__bf16*)smem;
  __bf16* xlds = (__bf16*)(smem + WLDS_BYTES);
  float*  dlds = (float*)(smem + WLDS_BYTES + XLDS_BYTES);
  float*  blds = (float*)(smem + WLDS_BYTES + XLDS_BYTES + DLDS_BYTES);

  const int tid = threadIdx.x;
  const int b   = blockIdx.z;
  const int h0  = blockIdx.y * 8;     // 8 output rows per workgroup
  const int w0  = blockIdx.x * 32;    // 32 output cols per workgroup

  // ---- stage weights as bf16, layout [tap][o][c] ----
  for (int idx = tid; idx < 9 * COUT * CIN; idx += 256) {
    int tap = idx >> 12;              // /4096
    int rem = idx & 4095;
    int o = rem >> 6, c = rem & 63;
    wlds[idx] = (__bf16)weight[o * (CIN * 9) + c * 9 + tap];
  }
  // ---- stage x halo tile (bf16), layout [hh][ww][c], zero-pad OOB ----
  for (int idx = tid; idx < XLDS_ELEMS; idx += 256) {
    int c  = idx / 340;
    int rem = idx - c * 340;
    int hh = rem / 34;
    int ww = rem - hh * 34;
    int h = h0 - 1 + hh, w = w0 - 1 + ww;
    float v = 0.0f;
    if ((unsigned)h < (unsigned)H_ && (unsigned)w < (unsigned)W_)
      v = x[((b * CIN + c) * H_ + h) * W_ + w];
    xlds[(hh * 34 + ww) * 64 + c] = (__bf16)v;
  }
  // ---- stage depth halo tile ----
  for (int idx = tid; idx < DLDS_ELEMS; idx += 256) {
    int hh = idx / 34, ww = idx - hh * 34;
    int h = h0 - 1 + hh, w = w0 - 1 + ww;
    dlds[idx] = ((unsigned)h < (unsigned)H_ && (unsigned)w < (unsigned)W_)
                    ? depth[(b * H_ + h) * W_ + w] : 0.0f;
  }
  if (tid < 64) blds[tid] = bias[tid];
  __syncthreads();

  const int lane  = tid & 31;
  const int wave  = tid >> 5;     // wave owns output row h0+wave
  const int n     = lane & 15;    // output column within 16-group / A row
  const int khalf = lane >> 4;

  v8f acc[2][4] = {};             // [pixel-group g][M-block t], 64x32 f32 tile

  // tap-invariant center depth
  float dc[2];
#pragma unroll
  for (int g = 0; g < 2; ++g)
    dc[g] = dlds[(wave + 1) * 34 + (g * 16 + n + 1)];

#pragma unroll
  for (int tap = 0; tap < 9; ++tap) {
    const int kh = tap / 3, kw = tap - kh * 3;   // constants after unroll
    const int hht = wave + kh;                   // tap row inside halo tile
    v16bf ssp[2];
    const __bf16* xcol[2];
#pragma unroll
    for (int g = 0; g < 2; ++g) {
      const int wwt = g * 16 + n + kw;           // tap pixel
      const float dt = dlds[hht * 34 + wwt];
      const __bf16 sb = (__bf16)__expf(-fabsf(dt - dc[g]));   // ALPHA = 1
#pragma unroll
      for (int i = 0; i < 16; ++i) ssp[g][i] = sb;            // splat
      xcol[g] = &xlds[(hht * 34 + wwt) * 64];
    }
#pragma unroll
    for (int kb = 0; kb < 64; kb += 32) {        // two K=32 chunks of C=64
      // ---- B fragments: two b128 LDS loads + packed bf16 scale ----
      v16bf Bf[2];
#pragma unroll
      for (int g = 0; g < 2; ++g) {
        const int cbase = kb + khalf * 16;       // 32B-aligned in LDS
        frag16 xf;
        xf.q[0] = *(const v4u*)(xcol[g] + cbase);
        xf.q[1] = *(const v4u*)(xcol[g] + cbase + 8);
        Bf[g] = xf.bf * ssp[g];                  // -> v_pk_mul_bf16 x8
      }
      // ---- A fragments (weights, 2 x b128) + WMMA chain ----
#pragma unroll
      for (int t = 0; t < 4; ++t) {
        const __bf16* wrow = &wlds[tap * 4096 + (t * 16 + n) * 64];
        // A 16x32 layout: element i -> K = khalf*16 + i
        frag16 wf;
        wf.q[0] = *(const v4u*)(wrow + kb + khalf * 16);       // elems 0..7
        wf.q[1] = *(const v4u*)(wrow + kb + khalf * 16 + 8);   // elems 8..15
        acc[0][t] = __builtin_amdgcn_wmma_f32_16x16x32_bf16(
            false, wf.bf, false, Bf[0], (short)0, acc[0][t], false, false);
        acc[1][t] = __builtin_amdgcn_wmma_f32_16x16x32_bf16(
            false, wf.bf, false, Bf[1], (short)0, acc[1][t], false, false);
      }
    }
  }

  // ---- epilogue: + bias, write 64x32 f32 tile ----
  const int hout = h0 + wave;
#pragma unroll
  for (int t = 0; t < 4; ++t) {
#pragma unroll
    for (int r = 0; r < 8; ++r) {
      const int o = t * 16 + r + 8 * khalf;   // C/D layout: M = r + 8*(lane>=16)
      const float bv = blds[o];
#pragma unroll
      for (int g = 0; g < 2; ++g) {
        out[((b * COUT + o) * H_ + hout) * W_ + (w0 + g * 16 + n)] =
            acc[g][t][r] + bv;
      }
    }
  }
}

extern "C" void kernel_launch(void* const* d_in, const int* in_sizes, int n_in,
                              void* d_out, int out_size, void* d_ws, size_t ws_size,
                              hipStream_t stream) {
  (void)in_sizes; (void)n_in; (void)out_size; (void)d_ws; (void)ws_size;
  const float* x      = (const float*)d_in[0];
  const float* depth  = (const float*)d_in[1];
  const float* weight = (const float*)d_in[2];
  const float* bias   = (const float*)d_in[3];
  float* out = (float*)d_out;

  dim3 grid(W_ / 32, H_ / 8, B_);   // (4, 16, 8) = 512 workgroups
  depthconv_wmma_bf16<<<grid, 256, SMEM_BYTES, stream>>>(x, depth, weight, bias, out);
}
